// RetrievalAugmentedProcessor_4234837754156
// MI455X (gfx1250) — matrix operand order repaired and checked by reading, not compiled, not run on hardware
//
#include <hip/hip_runtime.h>

typedef unsigned short u16;
typedef unsigned int   u32;
typedef __attribute__((ext_vector_type(16))) __bf16 bf16x16;
typedef __attribute__((ext_vector_type(8)))  float  floatx8;

#define DIM  1024
#define Bb   4
#define Ss   2048
#define Rr   3
#define Kk   256
#define MAXR 256
#define Hh   8
#define HD   128

// ---------------- helpers ----------------
__device__ __forceinline__ float bf2f(u16 h) { return __uint_as_float(((u32)h) << 16); }
__device__ __forceinline__ u16 f2bf(float f) {
    u32 u = __float_as_uint(f);
    u += 0x7fffu + ((u >> 16) & 1u);   // round-to-nearest-even
    return (u16)(u >> 16);
}
__device__ __forceinline__ float gelu_f(float x) {
    return 0.5f * x * (1.0f + erff(x * 0.70710678118654752f));
}
__device__ __forceinline__ float block_reduce_sum(float v, float* red) {
    int tid = threadIdx.x;
    red[tid] = v; __syncthreads();
    for (int off = 128; off > 0; off >>= 1) { if (tid < off) red[tid] += red[tid + off]; __syncthreads(); }
    float r = red[0]; __syncthreads(); return r;
}
__device__ __forceinline__ float block_reduce_max(float v, float* red) {
    int tid = threadIdx.x;
    red[tid] = v; __syncthreads();
    for (int off = 128; off > 0; off >>= 1) { if (tid < off) red[tid] = fmaxf(red[tid], red[tid + off]); __syncthreads(); }
    float r = red[0]; __syncthreads(); return r;
}

// async DMA: global -> LDS, 16 bytes per lane, tracked by ASYNCcnt (no VGPR staging)
__device__ __forceinline__ void async_g2l_b128(u32 ldsAddr, const void* gptr) {
    asm volatile("global_load_async_to_lds_b128 %0, %1, off" :: "v"(ldsAddr), "v"(gptr) : "memory");
}
__device__ __forceinline__ void wait_async0() {
    asm volatile("s_wait_asynccnt 0x0" ::: "memory");
}

// ---------------- fp32 -> bf16 convert ----------------
__global__ __launch_bounds__(256) void f2bf_kernel(const float* __restrict__ in, u16* __restrict__ out, long long n) {
    long long i = (long long)blockIdx.x * blockDim.x + threadIdx.x;
    long long stride = (long long)gridDim.x * blockDim.x;
    for (; i < n; i += stride) out[i] = f2bf(in[i]);
}

// ---------------- bf16 tile transpose: per z, in [Ss][DIM] -> out [DIM][Ss] ----------------
__global__ __launch_bounds__(256) void transpose_bf16_kernel(const u16* __restrict__ in, u16* __restrict__ out) {
    __shared__ u16 tile[32][33];
    int z = blockIdx.z;
    const u16* ip = in + (size_t)z * Ss * DIM;
    u16* op = out + (size_t)z * DIM * Ss;
    int s0 = blockIdx.x * 32, d0 = blockIdx.y * 32;
    int tx = threadIdx.x & 31, ty = threadIdx.x >> 5;  // 32 x 8
#pragma unroll
    for (int i = 0; i < 4; i++) { int s = ty + i * 8; tile[s][tx] = ip[(size_t)(s0 + s) * DIM + d0 + tx]; }
    __syncthreads();
#pragma unroll
    for (int i = 0; i < 4; i++) { int d = ty + i * 8; op[(size_t)(d0 + d) * Ss + s0 + tx] = tile[tx][d]; }
}

// ---------------- bf16 WMMA GEMM: out[M,N] = act(alpha*(A[M,K] @ B[K,N]) + bias[N]) ----------------
// Block tile 128x128x32, 8 waves (4 row x 2 col), wave tile 32x64 = 2x4 wmma 16x16x32.
// Double-buffered LDS filled by GLOBAL_LOAD_ASYNC_TO_LDS_B128 (ASYNCcnt), one barrier per K-step.
// Requires M%128==0, N%128==0, K%32==0 (true for all uses here).
union FragBF { uint4 q[2]; bf16x16 v; };

#define ASTR 40    // A LDS row stride (elems): 32 k + pad, rows 16B aligned
#define BSTR 144   // B LDS row stride (elems): 128 n + pad, rows 16B aligned

template<bool GACT, bool OBF>
__global__ __launch_bounds__(256)
void gemm_bf16_kernel(const u16* __restrict__ A, long long aB,
                      const u16* __restrict__ Bm, long long bB,
                      const float* __restrict__ bias, long long biasB,
                      void* __restrict__ Out, long long cB,
                      int M, int N, int K, float alpha)
{
    __shared__ u16 Asm[2][128 * ASTR];
    __shared__ u16 Bsm[2][32 * BSTR];
    int tid = threadIdx.x;
    int z = blockIdx.z;
    const u16* Ag = A + (size_t)z * aB;
    const u16* Bg = Bm + (size_t)z * bB;
    const float* bp = bias ? (bias + (size_t)z * biasB) : (const float*)nullptr;
    int mBase = blockIdx.y * 128;
    int nBase = blockIdx.x * 128;
    int wid = tid >> 5, lane = tid & 31;
    int wr = wid >> 1, wc = wid & 1;
    int lrow = lane & 15, hi = lane >> 4;

    floatx8 acc[2][4];
#pragma unroll
    for (int a0 = 0; a0 < 2; a0++)
#pragma unroll
        for (int a1 = 0; a1 < 4; a1++)
#pragma unroll
            for (int j = 0; j < 8; j++) acc[a0][a1][j] = 0.0f;

    // A: 128 rows x 32 k = 512 xb128; thread idx -> (row = idx>>2, quad = idx&3)
    // B: 32 rows x 128 n = 512 xb128; (row = idx>>4, quad = idx&15)
    // precomputed per-thread LDS byte offsets (generic-pointer low 32 bits = LDS address)
    int aR0 = tid >> 2, aQ = (tid & 3) * 8;
    int bR0 = tid >> 4, bQ = (tid & 15) * 8;
    u32 aL0 = (u32)(size_t)&Asm[0][aR0 * ASTR + aQ];
    u32 aL1 = (u32)(size_t)&Asm[0][(aR0 + 64) * ASTR + aQ];
    u32 bL0 = (u32)(size_t)&Bsm[0][bR0 * BSTR + bQ];
    u32 bL1 = (u32)(size_t)&Bsm[0][(bR0 + 16) * BSTR + bQ];
    u32 aBufStep = (u32)(128 * ASTR * 2);   // bytes between Asm[0] and Asm[1]
    u32 bBufStep = (u32)(32 * BSTR * 2);

    auto stage_async = [&](int kBase, int buf) {
        u32 ao = buf ? aBufStep : 0u;
        u32 bo = buf ? bBufStep : 0u;
        async_g2l_b128(aL0 + ao, Ag + (size_t)(mBase + aR0) * K + kBase + aQ);
        async_g2l_b128(aL1 + ao, Ag + (size_t)(mBase + aR0 + 64) * K + kBase + aQ);
        async_g2l_b128(bL0 + bo, Bg + (size_t)(kBase + bR0) * N + nBase + bQ);
        async_g2l_b128(bL1 + bo, Bg + (size_t)(kBase + bR0 + 16) * N + nBase + bQ);
    };

    int nk = K >> 5;
    stage_async(0, 0);
    wait_async0();
    __syncthreads();

    for (int kt = 0; kt < nk; kt++) {
        int cur = kt & 1;
        bool more = (kt + 1) < nk;
        if (more) stage_async((kt + 1) << 5, cur ^ 1);   // DMA next tile while we compute

        // A frag (16x32): lane<16 -> M=lane, K {0..7,16..23}; lane>=16 -> K {8..15,24..31}
        FragBF af[2], bfr[4];
#pragma unroll
        for (int tr = 0; tr < 2; tr++) {
            const u16* ar = &Asm[cur][(wr * 32 + tr * 16 + lrow) * ASTR];
            af[tr].q[0] = *(const uint4*)(ar + hi * 8);
            af[tr].q[1] = *(const uint4*)(ar + 16 + hi * 8);
        }
        // B frag (32x16): lane L -> N=L&15, K {0..7,16..23} (L<16) or {8..15,24..31} (L>=16)
#pragma unroll
        for (int tc = 0; tc < 4; tc++) {
            const u16* br = &Bsm[cur][wc * 64 + tc * 16 + lrow];
#pragma unroll
            for (int j = 0; j < 8; j++) {
                bfr[tc].v[j]     = __builtin_bit_cast(__bf16, br[(hi * 8 + j) * BSTR]);
                bfr[tc].v[8 + j] = __builtin_bit_cast(__bf16, br[(16 + hi * 8 + j) * BSTR]);
            }
        }
#pragma unroll
        for (int tr = 0; tr < 2; tr++)
#pragma unroll
            for (int tc = 0; tc < 4; tc++)
                acc[tr][tc] = __builtin_amdgcn_wmma_f32_16x16x32_bf16(
                    false, af[tr].v, false, bfr[tc].v, (short)0, acc[tr][tc], false, false);

        wait_async0();          // next buffer fully written (this wave's DMAs)
        __syncthreads();        // publish across waves; also guards reuse of cur
    }

    // C/D layout: lane L -> col = L&15; VGPR j -> row = j + 8*(L>>4)
    u16* outb = (u16*)Out + (size_t)z * cB;
    float* outf = (float*)Out + (size_t)z * cB;
#pragma unroll
    for (int tr = 0; tr < 2; tr++) {
#pragma unroll
        for (int tc = 0; tc < 4; tc++) {
            int gm0 = mBase + wr * 32 + tr * 16 + hi * 8;
            int gn = nBase + wc * 64 + tc * 16 + lrow;
            float bv = bp ? bp[gn] : 0.0f;
#pragma unroll
            for (int j = 0; j < 8; j++) {
                float v2 = acc[tr][tc][j] * alpha + bv;
                if (GACT) v2 = gelu_f(v2);
                if (OBF) outb[(size_t)(gm0 + j) * N + gn] = f2bf(v2);
                else     outf[(size_t)(gm0 + j) * N + gn] = v2;
            }
        }
    }
}

// ---------------- row softmax in place (n columns, one block per row) ----------------
__global__ __launch_bounds__(256) void softmax_rows_kernel(float* __restrict__ P, int n) {
    __shared__ float red[256];
    float* p = P + (size_t)blockIdx.x * n;
    int tid = threadIdx.x;
    float m = -3.0e38f;
    for (int c = tid; c < n; c += 256) m = fmaxf(m, p[c]);
    m = block_reduce_max(m, red);
    float s = 0.0f;
    for (int c = tid; c < n; c += 256) { float e = expf(p[c] - m); p[c] = e; s += e; }
    s = block_reduce_sum(s, red);
    float inv = 1.0f / s;
    for (int c = tid; c < n; c += 256) p[c] *= inv;
}

// ---------------- column sums of softmax matrix -> colw[b][t] ----------------
__global__ __launch_bounds__(256) void colsum_kernel(const float* __restrict__ P, float* __restrict__ colw) {
    int t = blockIdx.x * 256 + threadIdx.x;
    int b = blockIdx.y;
    const float* base = P + (size_t)b * Ss * Ss;
    float acc = 0.0f;
    for (int s = 0; s < Ss; s++) acc += base[(size_t)s * Ss + t];
    colw[b * Ss + t] = acc;
}

// ---------------- pooled[b][d] = (1/S) * sum_t colw[t] * qf[b][t][d] ----------------
__global__ __launch_bounds__(256) void pooled_kernel(const u16* __restrict__ qf_r, const float* __restrict__ colw,
                                                     float* __restrict__ pooled_r) {
    int d = blockIdx.x * 256 + threadIdx.x;
    int b = blockIdx.y;
    const u16* q = qf_r + (size_t)b * Ss * DIM;
    const float* cw = colw + b * Ss;
    float acc = 0.0f;
    for (int t = 0; t < Ss; t++) acc += cw[t] * bf2f(q[(size_t)t * DIM + d]);
    pooled_r[b * DIM + d] = acc * (1.0f / (float)Ss);
}

// ---------------- LN(pooled) @ pool_w + pool_b -> queries (one block per (r,b)) ----------------
__global__ __launch_bounds__(256) void pool_ln_linear_kernel(const float* __restrict__ pooled,
                                                             const float* __restrict__ g, const float* __restrict__ bta,
                                                             const float* __restrict__ poolw, const float* __restrict__ poolb,
                                                             float* __restrict__ queries) {
    __shared__ float ln[DIM];
    __shared__ float red[256];
    int rb = blockIdx.x; int r = rb >> 2; int tid = threadIdx.x;
    const float* pr = pooled + (size_t)rb * DIM;
    float s = 0.0f;
    for (int d = tid; d < DIM; d += 256) s += pr[d];
    float mean = block_reduce_sum(s, red) * (1.0f / DIM);
    float v = 0.0f;
    for (int d = tid; d < DIM; d += 256) { float t = pr[d] - mean; v += t * t; }
    float var = block_reduce_sum(v, red) * (1.0f / DIM);
    float inv = rsqrtf(var + 1e-5f);
    for (int d = tid; d < DIM; d += 256) ln[d] = (pr[d] - mean) * inv * g[r * DIM + d] + bta[r * DIM + d];
    __syncthreads();
    const float* W = poolw + (size_t)r * DIM * DIM;
    for (int n = tid; n < DIM; n += 256) {
        float acc = 0.0f;
        for (int d = 0; d < DIM; d++) acc += ln[d] * W[(size_t)d * DIM + n];
        queries[(size_t)rb * DIM + n] = acc + poolb[r * DIM + n];
    }
}

// ---------------- qpart[rb][n] = queries[rb] . rel_w1[DIM:2*DIM, n] ----------------
__global__ __launch_bounds__(256) void qpart_kernel(const float* __restrict__ queries, const float* __restrict__ relw1,
                                                    float* __restrict__ qpart) {
    __shared__ float qs[DIM];
    int rb = blockIdx.x; int tid = threadIdx.x;
    for (int d = tid; d < DIM; d += 256) qs[d] = queries[(size_t)rb * DIM + d];
    __syncthreads();
    for (int n = tid; n < 512; n += 256) {
        float acc = 0.0f;
        for (int d = 0; d < DIM; d++) acc += qs[d] * relw1[(size_t)(DIM + d) * 512 + n];
        qpart[rb * 512 + n] = acc;
    }
}

// ---------------- h = GELU(hraw + qpart + rel_b1) in place ----------------
__global__ __launch_bounds__(256) void relact_kernel(float* __restrict__ h, const float* __restrict__ qpart,
                                                     const float* __restrict__ relb1) {
    size_t i = (size_t)blockIdx.x * 256 + threadIdx.x;
    int n = (int)(i & 511);
    int rb = (int)(i >> 17);  // / (512*256)
    float v = h[i] + qpart[rb * 512 + n] + relb1[n];
    h[i] = gelu_f(v);
}

// ---------------- rel[row] = sigmoid(h[row] . rel_w2 + rel_b2), one wave32 per row ----------------
__global__ __launch_bounds__(256) void relscore_kernel(const float* __restrict__ h, const float* __restrict__ w2,
                                                       const float* __restrict__ b2, float* __restrict__ rel) {
    int row = blockIdx.x * 8 + (threadIdx.x >> 5);
    int lane = threadIdx.x & 31;
    const float* hr = h + (size_t)row * 512;
    float acc = 0.0f;
    for (int n = lane; n < 512; n += 32) acc += hr[n] * w2[n];
    for (int off = 16; off > 0; off >>= 1) acc += __shfl_xor(acc, off);
    if (lane == 0) rel[row] = 1.0f / (1.0f + expf(-(acc + b2[0])));
}

// ---------------- threshold + exact top-MAXR via rank counting (one block per batch) ----------------
__global__ __launch_bounds__(256) void topk_kernel(const float* __restrict__ rel, int* __restrict__ selidx,
                                                   int* __restrict__ keyvalid, int* __restrict__ anyvalid) {
    __shared__ float sm[Rr * Kk];
    __shared__ int anyf;
    int b = blockIdx.x, tid = threadIdx.x;
    if (tid == 0) anyf = 0;
    for (int j = tid; j < Rr * Kk; j += 256) {
        int r = j >> 8, k = j & 255;
        float v = rel[((size_t)r * Bb + b) * Kk + k];
        sm[j] = (v >= 0.5f) ? v : -__builtin_inff();
    }
    __syncthreads();
    int la = 0;
    for (int j = tid; j < Rr * Kk; j += 256) {
        float vj = sm[j];
        int cnt = 0;
        for (int i2 = 0; i2 < Rr * Kk; i2++) {
            float vi = sm[i2];
            cnt += (vi > vj) || (vi == vj && i2 < j);
        }
        if (cnt < MAXR) {
            selidx[b * MAXR + cnt] = j;
            int val = (vj > -3.0e38f) ? 1 : 0;
            keyvalid[b * MAXR + cnt] = val;
            la |= val;
        }
    }
    if (la) atomicOr(&anyf, 1);
    __syncthreads();
    if (tid == 0) anyvalid[b] = anyf;
}

// ---------------- gather selected tokens -> bf16 ----------------
__global__ __launch_bounds__(256) void gather_kernel(const float* __restrict__ retrieved, const int* __restrict__ selidx,
                                                     u16* __restrict__ seltokb) {
    size_t i = (size_t)blockIdx.x * 256 + threadIdx.x;
    int d = (int)(i & (DIM - 1));
    size_t bs = i >> 10;
    int slot = (int)(bs & (MAXR - 1));
    int b = (int)(bs >> 8);
    int j = selidx[b * MAXR + slot];
    int r = j >> 8, k = j & 255;
    seltokb[i] = f2bf(retrieved[(((size_t)r * Bb + b) * Kk + k) * DIM + d]);
}

// ---------------- fusion cross-attention: one wave32 per (b,h,s) query row ----------------
__global__ __launch_bounds__(256) void fusion_attn_kernel(const u16* __restrict__ qb, const u16* __restrict__ kb,
                                                          const u16* __restrict__ vb, const int* __restrict__ keyvalid,
                                                          u16* __restrict__ outb) {
    __shared__ float qrow[8][HD];
    __shared__ float aw[8][MAXR];
    int w = threadIdx.x >> 5, lane = threadIdx.x & 31;
    int gw = blockIdx.x * 8 + w;
    int s = gw & (Ss - 1);
    int bh = gw >> 11;
    int h = bh & (Hh - 1);
    int b = bh >> 3;
    const u16* qp = qb + ((size_t)b * Ss + s) * DIM + h * HD;
#pragma unroll
    for (int i = 0; i < 4; i++) { int d = lane + i * 32; qrow[w][d] = bf2f(qp[d]); }
    __syncthreads();
    const u16* kbase = kb + (size_t)b * MAXR * DIM + h * HD;
    float sc[8];
#pragma unroll
    for (int i = 0; i < 8; i++) {
        int t = lane + i * 32;
        const u16* kr = kbase + (size_t)t * DIM;
        float acc = 0.0f;
        for (int d = 0; d < HD; d++) acc += qrow[w][d] * bf2f(kr[d]);
        sc[i] = keyvalid[b * MAXR + t] ? acc * 0.088388347648318447f : -1e9f;  // 1/sqrt(128)
    }
    float m = sc[0];
#pragma unroll
    for (int i = 1; i < 8; i++) m = fmaxf(m, sc[i]);
    for (int off = 16; off > 0; off >>= 1) m = fmaxf(m, __shfl_xor(m, off));
    float sum = 0.0f;
#pragma unroll
    for (int i = 0; i < 8; i++) { sc[i] = expf(sc[i] - m); sum += sc[i]; }
    for (int off = 16; off > 0; off >>= 1) sum += __shfl_xor(sum, off);
    float inv = 1.0f / sum;
#pragma unroll
    for (int i = 0; i < 8; i++) aw[w][lane + i * 32] = sc[i] * inv;
    __syncthreads();
    const u16* vbase = vb + (size_t)b * MAXR * DIM + h * HD;
    u16* op = outb + ((size_t)b * Ss + s) * DIM + h * HD;
#pragma unroll
    for (int c = 0; c < 4; c++) {
        int d = lane + c * 32;
        float acc = 0.0f;
        for (int t = 0; t < MAXR; t++) acc += aw[w][t] * bf2f(vbase[(size_t)t * DIM + d]);
        op[d] = f2bf(acc);
    }
}

// ---------------- final: out = anyvalid ? LN(x + fused)*g+b : x (one block per row) ----------------
__global__ __launch_bounds__(256) void final_ln_kernel(const float* __restrict__ x, const float* __restrict__ fused,
                                                       const float* __restrict__ g, const float* __restrict__ be,
                                                       const int* __restrict__ anyvalid, float* __restrict__ out) {
    __shared__ float y[DIM];
    __shared__ float red[256];
    size_t row = blockIdx.x;
    int b = (int)(row >> 11);
    int tid = threadIdx.x;
    const float* xr = x + row * DIM;
    const float* fr = fused + row * DIM;
    float s = 0.0f;
    for (int d = tid; d < DIM; d += 256) { float t = xr[d] + fr[d]; y[d] = t; s += t; }
    float mean = block_reduce_sum(s, red) * (1.0f / DIM);
    float v = 0.0f;
    for (int d = tid; d < DIM; d += 256) { float t = y[d] - mean; v += t * t; }
    float var = block_reduce_sum(v, red) * (1.0f / DIM);
    float inv = rsqrtf(var + 1e-5f);
    int av = anyvalid[b];
    float* orow = out + row * DIM;
    for (int d = tid; d < DIM; d += 256) {
        float e = (y[d] - mean) * inv * g[d] + be[d];
        orow[d] = av ? e : xr[d];
    }
}

// ==================== host ====================
extern "C" void kernel_launch(void* const* d_in, const int* in_sizes, int n_in,
                              void* d_out, int out_size, void* d_ws, size_t ws_size,
                              hipStream_t stream) {
    (void)in_sizes; (void)n_in; (void)out_size; (void)ws_size;
    const float* x         = (const float*)d_in[0];
    const float* retrieved = (const float*)d_in[1];
    const float* qg_w1     = (const float*)d_in[2];
    const float* qg_b1     = (const float*)d_in[3];
    const float* qg_w2     = (const float*)d_in[4];
    const float* qg_b2     = (const float*)d_in[5];
    const float* pln_g     = (const float*)d_in[6];
    const float* pln_b     = (const float*)d_in[7];
    const float* pool_w    = (const float*)d_in[8];
    const float* pool_b    = (const float*)d_in[9];
    const float* rel_w1    = (const float*)d_in[10];
    const float* rel_b1    = (const float*)d_in[11];
    const float* rel_w2    = (const float*)d_in[12];
    const float* rel_b2    = (const float*)d_in[13];
    const float* in_w      = (const float*)d_in[14];
    const float* in_b      = (const float*)d_in[15];
    const float* out_w     = (const float*)d_in[16];
    const float* out_b     = (const float*)d_in[17];
    const float* fln_g     = (const float*)d_in[18];
    const float* fln_b     = (const float*)d_in[19];
    float* out = (float*)d_out;

    char* base = (char*)d_ws;
    size_t off = 0;
    auto alloc = [&](size_t bytes) -> char* {
        char* p = base + off;
        off = (off + bytes + 255) & ~(size_t)255;
        return p;
    };
    const size_t BSD = (size_t)Bb * Ss * DIM;

    u16*   xb      = (u16*)alloc(BSD * 2);
    u16*   w1b     = (u16*)alloc((size_t)Rr * DIM * DIM * 2);
    u16*   w2b     = (u16*)alloc((size_t)Rr * DIM * DIM * 2);
    u16*   qf1b    = (u16*)alloc((size_t)Rr * BSD * 2);
    u16*   qfb     = (u16*)alloc((size_t)Rr * BSD * 2);
    float* scores  = (float*)alloc((size_t)Bb * Ss * Ss * 4);
    float* colw    = (float*)alloc((size_t)Bb * Ss * 4);
    float* pooled  = (float*)alloc((size_t)Rr * Bb * DIM * 4);
    float* queries = (float*)alloc((size_t)Rr * Bb * DIM * 4);
    u16*   retb    = (u16*)alloc((size_t)Rr * Bb * Kk * DIM * 2);
    u16*   rw1b    = (u16*)alloc((size_t)2 * DIM * 512 * 2);
    float* qpart   = (float*)alloc((size_t)Rr * Bb * 512 * 4);
    float* hraw    = (float*)alloc((size_t)Rr * Bb * Kk * 512 * 4);
    float* relv    = (float*)alloc((size_t)Rr * Bb * Kk * 4);
    int*   selidx  = (int*)alloc((size_t)Bb * MAXR * 4);
    int*   keyval  = (int*)alloc((size_t)Bb * MAXR * 4);
    int*   anyval  = (int*)alloc(Bb * 4);
    u16*   seltokb = (u16*)alloc((size_t)Bb * MAXR * DIM * 2);
    u16*   inwb    = (u16*)alloc((size_t)3 * DIM * DIM * 2);
    u16*   outwb   = (u16*)alloc((size_t)DIM * DIM * 2);
    u16*   qb2     = (u16*)alloc(BSD * 2);
    u16*   kb2     = (u16*)alloc((size_t)Bb * MAXR * DIM * 2);
    u16*   vb2     = (u16*)alloc((size_t)Bb * MAXR * DIM * 2);
    u16*   qfT     = qf1b;            // alias: qf1b dead after qf GEMM; qfT dead after pooling loop
    u16*   attnb   = qf1b;            // alias: used after qfT is dead
    float* fusedf  = scores;          // alias: scores dead after pooling loop

    // ---- fp32 -> bf16 conversions ----
    f2bf_kernel<<<2048, 256, 0, stream>>>(x, xb, (long long)BSD);
    f2bf_kernel<<<2048, 256, 0, stream>>>(qg_w1, w1b, (long long)Rr * DIM * DIM);
    f2bf_kernel<<<2048, 256, 0, stream>>>(qg_w2, w2b, (long long)Rr * DIM * DIM);
    f2bf_kernel<<<2048, 256, 0, stream>>>(retrieved, retb, (long long)Rr * Bb * Kk * DIM);
    f2bf_kernel<<<2048, 256, 0, stream>>>(rel_w1, rw1b, (long long)2 * DIM * 512);
    f2bf_kernel<<<2048, 256, 0, stream>>>(in_w, inwb, (long long)3 * DIM * DIM);
    f2bf_kernel<<<2048, 256, 0, stream>>>(out_w, outwb, (long long)DIM * DIM);

    // ---- query generation MLP (batched over R) ----
    {
        dim3 g(DIM / 128, (Bb * Ss) / 128, Rr);
        gemm_bf16_kernel<true, true><<<g, 256, 0, stream>>>(
            xb, 0, w1b, (long long)DIM * DIM, qg_b1, DIM, qf1b, (long long)BSD, Bb * Ss, DIM, DIM, 1.0f);
        gemm_bf16_kernel<false, true><<<g, 256, 0, stream>>>(
            qf1b, (long long)BSD, w2b, (long long)DIM * DIM, qg_b2, DIM, qfb, (long long)BSD, Bb * Ss, DIM, DIM, 1.0f);
    }

    // ---- materialize qf^T (per (r,b): [DIM][Ss]) into dead qf1 buffer ----
    transpose_bf16_kernel<<<dim3(Ss / 32, DIM / 32, Rr * Bb), 256, 0, stream>>>(qfb, qfT);

    // ---- self-attn pooling per retriever ----
    for (int r = 0; r < Rr; r++) {
        const u16* qr = qfb + (size_t)r * BSD;
        const u16* qTr = qfT + (size_t)r * BSD;
        {
            dim3 g(Ss / 128, Ss / 128, Bb);
            gemm_bf16_kernel<false, false><<<g, 256, 0, stream>>>(
                qr, (long long)Ss * DIM, qTr, (long long)DIM * Ss, (const float*)nullptr, 0,
                scores, (long long)Ss * Ss, Ss, Ss, DIM, 0.03125f /*1/sqrt(1024)*/);
        }
        softmax_rows_kernel<<<Bb * Ss, 256, 0, stream>>>(scores, Ss);
        colsum_kernel<<<dim3(Ss / 256, Bb), 256, 0, stream>>>(scores, colw);
        pooled_kernel<<<dim3(DIM / 256, Bb), 256, 0, stream>>>(qr, colw, pooled + (size_t)r * Bb * DIM);
    }
    pool_ln_linear_kernel<<<Rr * Bb, 256, 0, stream>>>(pooled, pln_g, pln_b, pool_w, pool_b, queries);

    // ---- relevance scoring ----
    qpart_kernel<<<Rr * Bb, 256, 0, stream>>>(queries, rel_w1, qpart);
    {
        dim3 g(512 / 128, (Rr * Bb * Kk) / 128, 1);
        gemm_bf16_kernel<false, false><<<g, 256, 0, stream>>>(
            retb, 0, rw1b, 0, (const float*)nullptr, 0, hraw, 0, Rr * Bb * Kk, 512, DIM, 1.0f);
    }
    relact_kernel<<<(Rr * Bb * Kk * 512) / 256, 256, 0, stream>>>(hraw, qpart, rel_b1);
    relscore_kernel<<<(Rr * Bb * Kk) / 8, 256, 0, stream>>>(hraw, rel_w2, rel_b2, relv);

    // ---- threshold + top-k + gather ----
    topk_kernel<<<Bb, 256, 0, stream>>>(relv, selidx, keyval, anyval);
    gather_kernel<<<(Bb * MAXR * DIM) / 256, 256, 0, stream>>>(retrieved, selidx, seltokb);

    // ---- fusion q/k/v projections ----
    {
        dim3 g(DIM / 128, (Bb * Ss) / 128, 1);
        gemm_bf16_kernel<false, true><<<g, 256, 0, stream>>>(
            xb, 0, inwb, 0, in_b, 0, qb2, 0, Bb * Ss, DIM, DIM, 1.0f);
    }
    {
        dim3 g(DIM / 128, (Bb * MAXR) / 128, 1);
        gemm_bf16_kernel<false, true><<<g, 256, 0, stream>>>(
            seltokb, 0, inwb + (size_t)DIM * DIM, 0, in_b + DIM, 0, kb2, 0, Bb * MAXR, DIM, DIM, 1.0f);
        gemm_bf16_kernel<false, true><<<g, 256, 0, stream>>>(
            seltokb, 0, inwb + (size_t)2 * DIM * DIM, 0, in_b + 2 * DIM, 0, vb2, 0, Bb * MAXR, DIM, DIM, 1.0f);
    }

    // ---- cross attention + out proj + residual LN ----
    fusion_attn_kernel<<<(Bb * Hh * Ss) / 8, 256, 0, stream>>>(qb2, kb2, vb2, keyval, attnb);
    {
        dim3 g(DIM / 128, (Bb * Ss) / 128, 1);
        gemm_bf16_kernel<false, false><<<g, 256, 0, stream>>>(
            attnb, 0, outwb, 0, out_b, 0, fusedf, 0, Bb * Ss, DIM, DIM, 1.0f);
    }
    final_ln_kernel<<<Bb * Ss, 256, 0, stream>>>(x, fusedf, fln_g, fln_b, anyval, out);
}
